// SoftHardRouteBlock_38130719654140
// MI455X (gfx1250) — compile-verified, matrix operand order checked by reading, not run on hardware
//
#include <hip/hip_runtime.h>

#define B_  8
#define N_  4096
#define D_  256
#define QK_ 128
#define M_  32

typedef __attribute__((ext_vector_type(16))) __bf16 bf16x16;
typedef __attribute__((ext_vector_type(8)))  __bf16 bf16x8;
typedef __attribute__((ext_vector_type(8)))  float  f32x8;
typedef __attribute__((ext_vector_type(4)))  int    v4i;

static __device__ __forceinline__ f32x8 wmma_bf16(bf16x16 a, bf16x16 b, f32x8 c) {
  // D = A(16x32 bf16) * B(32x16 bf16) + C(16x16 f32)
  return __builtin_amdgcn_wmma_f32_16x16x32_bf16(false, a, false, b, (short)0, c,
                                                 false, false);
}

// Load a wave32 16-bit operand fragment from a row-major row pointer:
// elements [0..7] and [16..23] relative to p (p already includes the per-lane
// khalf offset).  Two 16-byte loads.
static __device__ __forceinline__ bf16x16 load_frag_contig(const __bf16* p) {
  const bf16x8* v = (const bf16x8*)p;
  bf16x8 lo = v[0];   // K = +0..+7
  bf16x8 hi = v[2];   // K = +16..+23
  return __builtin_shufflevector(lo, hi, 0,1,2,3,4,5,6,7,8,9,10,11,12,13,14,15);
}

// ---- CDNA5 async global->LDS staging (ASYNCcnt-tracked), with safe fallback
#if __has_builtin(__builtin_amdgcn_global_load_async_to_lds_b128)
#define HAVE_ASYNC_LDS 1
typedef __attribute__((address_space(1))) v4i g_v4i_t;   // HIP prints as __device__
typedef __attribute__((address_space(3))) v4i l_v4i_t;   // LDS
#else
#define HAVE_ASYNC_LDS 0
#endif

static __device__ __forceinline__ void stage16(void* lds_dst, const void* gsrc) {
#if HAVE_ASYNC_LDS
  // emits GLOBAL_LOAD_ASYNC_TO_LDS_B128 (VGLOBAL op 98): per-lane 16B move,
  // no VGPR data path, tracked with ASYNCcnt.
  __builtin_amdgcn_global_load_async_to_lds_b128((g_v4i_t*)gsrc,
                                                 (l_v4i_t*)lds_dst,
                                                 /*imm offset*/0, /*cpol*/0);
#else
  *(float4*)lds_dst = *(const float4*)gsrc;
#endif
}

static __device__ __forceinline__ void wait_async_all() {
#if HAVE_ASYNC_LDS
#if __has_builtin(__builtin_amdgcn_s_wait_asynccnt)
  __builtin_amdgcn_s_wait_asynccnt(0);
#else
  asm volatile("s_wait_asynccnt 0" ::: "memory");
#endif
#endif
}

// ---------------------------------------------------------------------------
// Kernel 1: Q = H0*Wq, K = X0*Wk, output bf16.  One wave per 16x16 output
// tile, K-dim D=256 in 8 WMMA steps.  f32 inputs converted to bf16 inline.
// ---------------------------------------------------------------------------
__global__ __launch_bounds__(256) void proj_kernel(
    const float* __restrict__ H0, const float* __restrict__ X0,
    const float* __restrict__ Wq, const float* __restrict__ Wk,
    __bf16* __restrict__ Qb, __bf16* __restrict__ Kb)
{
  int wave = blockIdx.x * 8 + (threadIdx.x >> 5);
  int lane = threadIdx.x & 31;
  int which   = wave >> 14;      // 0 -> Q (H0,Wq), 1 -> K (X0,Wk)
  int t       = wave & 16383;
  int qt      = t & 7;           // QK tile (8 tiles of 16)
  int rowTile = t >> 3;          // global token tile over B*N (2048 tiles)
  const float* src = which ? X0 : H0;
  const float* W   = which ? Wk : Wq;
  __bf16*      dst = which ? Kb : Qb;

  int r     = lane & 15;
  int khalf = (lane >> 4) * 8;   // 0 or 8 (ISA wave32 16-bit operand striping)
  int row0  = rowTile * 16;

  f32x8 c = {};
  const float* arow = src + (size_t)(row0 + r) * D_;
  const float* bcol = W + qt * 16 + r;
#pragma unroll 2
  for (int k0 = 0; k0 < D_; k0 += 32) {
    bf16x16 a, bm;
    const float* ap = arow + k0 + khalf;
    const float* bp = bcol + (size_t)(k0 + khalf) * QK_;
#pragma unroll
    for (int j = 0; j < 8; ++j) {
      a[j]    = (__bf16)ap[j];
      a[j+8]  = (__bf16)ap[16 + j];
      bm[j]   = (__bf16)bp[(size_t)j * QK_];
      bm[j+8] = (__bf16)bp[(size_t)(16 + j) * QK_];
    }
    c = wmma_bf16(a, bm, c);
  }
  // C layout: VGPR vr holds row (vr + khalf), col = lane&15
#pragma unroll
  for (int vr = 0; vr < 8; ++vr)
    dst[(size_t)(row0 + khalf + vr) * QK_ + qt * 16 + r] = (__bf16)c[vr];
}

// ---------------------------------------------------------------------------
// Kernel 2: fused S = Q*K^T (bf16 WMMA, f32 acc) + streaming sum(exp) ->
// scores = logsumexp(S, axis=-1).  S (512 MB) is never materialized.
// Block = 8 waves sharing one K stream: K chunks (64 rows x 256 B = 16 KB,
// contiguous) are double-buffered into LDS with async-to-LDS copies so the
// prefetch of chunk i+1 overlaps the 16 WMMAs + 32 exps of chunk i, and L2
// read traffic drops 8x vs per-wave streaming.  |S| < ~0.5 by construction
// (0.02-scaled weights), so a shift-free exp-sum is numerically safe.
// ---------------------------------------------------------------------------
__global__ __launch_bounds__(256) void scores_kernel(
    const __bf16* __restrict__ Qb, const __bf16* __restrict__ Kb,
    float* __restrict__ scores)
{
  __shared__ __attribute__((aligned(16))) char smem[2][16384];

  int wid  = threadIdx.x >> 5;
  int lane = threadIdx.x & 31;
  int b    = blockIdx.x >> 5;      // 32 blocks per batch
  int ntg  = blockIdx.x & 31;      // group of 8 query tiles
  int nt   = ntg * 8 + wid;        // this wave's query tile
  int r     = lane & 15;
  int khalf = (lane >> 4) * 8;
  const float scale = 0.08838834764831845f;  // QK^-0.5

  // resident A fragments: 16 query rows x full QK=128
  bf16x16 aq[4];
  const __bf16* qrow = Qb + ((size_t)b * N_ + nt * 16 + r) * QK_;
#pragma unroll
  for (int kc = 0; kc < 4; ++kc)
    aq[kc] = load_frag_contig(qrow + kc * 32 + khalf);

  const char* kbytes = (const char*)(Kb + (size_t)b * N_ * QK_);

  // stage chunk 0 (each thread moves 4 x 16B of the contiguous 16KB block)
#pragma unroll
  for (int i = 0; i < 4; ++i) {
    int off = (threadIdx.x + 256 * i) * 16;
    stage16(&smem[0][off], kbytes + off);
  }
  wait_async_all();
  __syncthreads();

  float se[8] = {0.f,0.f,0.f,0.f,0.f,0.f,0.f,0.f};
  const int nchunks = N_ / 64;
  for (int it = 0; it < nchunks; ++it) {
    int cur = it & 1;
    if (it + 1 < nchunks) {           // async prefetch of next chunk
      const char* g = kbytes + (size_t)(it + 1) * 16384;
#pragma unroll
      for (int i = 0; i < 4; ++i) {
        int off = (threadIdx.x + 256 * i) * 16;
        stage16(&smem[cur ^ 1][off], g + off);
      }
    }
    const __bf16* kl = (const __bf16*)smem[cur];
#pragma unroll
    for (int mt = 0; mt < 4; ++mt) {  // 4 key tiles per chunk
      const __bf16* krow = kl + (size_t)(mt * 16 + r) * QK_;
      f32x8 c = {};
#pragma unroll
      for (int kc = 0; kc < 4; ++kc) {
        bf16x16 bm = load_frag_contig(krow + kc * 32 + khalf);
        c = wmma_bf16(aq[kc], bm, c);
      }
#pragma unroll
      for (int vr = 0; vr < 8; ++vr)
        se[vr] += __expf(c[vr] * scale);   // TRANS pipe, co-executes with WMMA
    }
    wait_async_all();   // our prefetch landed
    __syncthreads();    // everyone done reading cur, next buffer published
  }

  // reduce over the 16 lanes of each half-wave (each half holds one row set)
#pragma unroll
  for (int vr = 0; vr < 8; ++vr) {
    float s = se[vr];
    s += __shfl_xor(s, 1, 32);
    s += __shfl_xor(s, 2, 32);
    s += __shfl_xor(s, 4, 32);
    s += __shfl_xor(s, 8, 32);
    if (r == 0)
      scores[(size_t)b * N_ + nt * 16 + khalf + vr] = __logf(s);
  }
}

// ---------------------------------------------------------------------------
// Kernel 3: generic row softmax (w_keep over scores; slot_w over slot_logits)
// ---------------------------------------------------------------------------
__global__ __launch_bounds__(256) void softmax_rows_kernel(
    const float* __restrict__ in, float* __restrict__ out, int cols)
{
  __shared__ float red[256];
  int row = blockIdx.x;
  int tid = threadIdx.x;
  const float* p = in + (size_t)row * cols;
  float mx = -3.4e38f;
  for (int i = tid; i < cols; i += 256) mx = fmaxf(mx, p[i]);
  red[tid] = mx; __syncthreads();
  for (int s = 128; s > 0; s >>= 1) {
    if (tid < s) red[tid] = fmaxf(red[tid], red[tid + s]);
    __syncthreads();
  }
  float m = red[0]; __syncthreads();
  float sum = 0.f;
  for (int i = tid; i < cols; i += 256) sum += __expf(p[i] - m);
  red[tid] = sum; __syncthreads();
  for (int s = 128; s > 0; s >>= 1) {
    if (tid < s) red[tid] += red[tid + s];
    __syncthreads();
  }
  float inv = 1.0f / red[0];
  float* o = out + (size_t)row * cols;
  for (int i = tid; i < cols; i += 256) o[i] = __expf(p[i] - m) * inv;
}

// ---------------------------------------------------------------------------
// Kernel 4: per-(b,m) denominator + normalized bf16 coefficient matrix
// coeff[b,m,n] = slot_w[m,n]*w_keep[b,n] / (sum_n(...) + 1e-6)
// ---------------------------------------------------------------------------
__global__ __launch_bounds__(256) void coeff_kernel(
    const float* __restrict__ slot_w, const float* __restrict__ w_keep,
    __bf16* __restrict__ coeffb)
{
  __shared__ float red[256];
  int b = blockIdx.x >> 5;
  int m = blockIdx.x & 31;
  int tid = threadIdx.x;
  const float* sw = slot_w + (size_t)m * N_;
  const float* wk = w_keep + (size_t)b * N_;
  float s = 0.f;
  for (int i = tid; i < N_; i += 256) s += sw[i] * wk[i];
  red[tid] = s; __syncthreads();
  for (int st = 128; st > 0; st >>= 1) {
    if (tid < st) red[tid] += red[tid + st];
    __syncthreads();
  }
  float inv = 1.0f / (red[0] + 1e-6f);
  __bf16* o = coeffb + ((size_t)b * M_ + m) * N_;
  for (int i = tid; i < N_; i += 256) o[i] = (__bf16)(sw[i] * wk[i] * inv);
}

// ---------------------------------------------------------------------------
// Kernel 5: out[b,m,d] = sum_n coeff[b,m,n] * H0[b,n,d]  (bf16 WMMA).
// n-dim split 8 ways for parallelism; partial 16x16 tiles accumulated into
// pre-zeroed d_out with global f32 atomics.
// ---------------------------------------------------------------------------
__global__ __launch_bounds__(256) void out_kernel(
    const __bf16* __restrict__ coeffb, const float* __restrict__ H0,
    float* __restrict__ out)
{
  int wave = blockIdx.x * 8 + (threadIdx.x >> 5);
  int lane = threadIdx.x & 31;
  int ns = wave & 7;             // n split (512 each)
  int dt = (wave >> 3) & 15;     // d tile (16 of 16)
  int mt = (wave >> 7) & 1;      // m tile (2 of 16)
  int b  = wave >> 8;
  int r     = lane & 15;
  int khalf = (lane >> 4) * 8;

  const __bf16* arow = coeffb + ((size_t)b * M_ + mt * 16 + r) * N_;
  const float*  hb   = H0 + (size_t)b * N_ * D_ + dt * 16 + r;

  f32x8 c = {};
  int n0beg = ns * 512;
  for (int n0 = n0beg; n0 < n0beg + 512; n0 += 32) {
    bf16x16 a = load_frag_contig(arow + n0 + khalf);
    bf16x16 bm;
    const float* hp = hb + (size_t)(n0 + khalf) * D_;
#pragma unroll
    for (int j = 0; j < 8; ++j) {
      bm[j]   = (__bf16)hp[(size_t)j * D_];
      bm[j+8] = (__bf16)hp[(size_t)(16 + j) * D_];
    }
    c = wmma_bf16(a, bm, c);
  }
  int mrow = mt * 16 + khalf;
#pragma unroll
  for (int vr = 0; vr < 8; ++vr)
    atomicAdd(&out[((size_t)b * M_ + mrow + vr) * D_ + dt * 16 + r], c[vr]);
}

// ---------------------------------------------------------------------------
extern "C" void kernel_launch(void* const* d_in, const int* in_sizes, int n_in,
                              void* d_out, int out_size, void* d_ws, size_t ws_size,
                              hipStream_t stream)
{
  const float* X0 = (const float*)d_in[0];
  const float* H0 = (const float*)d_in[1];
  const float* Wq = (const float*)d_in[2];
  const float* Wk = (const float*)d_in[3];
  const float* slot_logits = (const float*)d_in[4];
  float* out = (float*)d_out;

  // workspace carve-up (~19 MB total)
  char* ws = (char*)d_ws;
  __bf16* Qb = (__bf16*)ws;        ws += (size_t)B_ * N_ * QK_ * 2;  // 8 MB
  __bf16* Kb = (__bf16*)ws;        ws += (size_t)B_ * N_ * QK_ * 2;  // 8 MB
  float* scores = (float*)ws;      ws += (size_t)B_ * N_ * 4;        // 128 KB
  float* w_keep = (float*)ws;      ws += (size_t)B_ * N_ * 4;        // 128 KB
  float* slot_w = (float*)ws;      ws += (size_t)M_ * N_ * 4;        // 512 KB
  __bf16* coeffb = (__bf16*)ws;    ws += (size_t)B_ * M_ * N_ * 2;   // 2 MB
  (void)in_sizes; (void)n_in; (void)ws_size;

  (void)hipMemsetAsync(d_out, 0, (size_t)out_size * sizeof(float), stream);

  // 32768 tile-waves: 2 matrices x 2048 token tiles x 8 QK tiles
  proj_kernel<<<4096, 256, 0, stream>>>(H0, X0, Wq, Wk, Qb, Kb);
  // 256 blocks: 8 batches x 32 query-tile groups (8 waves each)
  scores_kernel<<<256, 256, 0, stream>>>(Qb, Kb, scores);
  softmax_rows_kernel<<<B_, 256, 0, stream>>>(scores, w_keep, N_);
  softmax_rows_kernel<<<M_, 256, 0, stream>>>(slot_logits, slot_w, N_);
  coeff_kernel<<<B_ * M_, 256, 0, stream>>>(slot_w, w_keep, coeffb);
  // 2048 waves: 8 batches x 2 m-tiles x 16 d-tiles x 8 n-splits
  out_kernel<<<256, 256, 0, stream>>>(coeffb, H0, out);
}
